// VQVAE3D_27857157881934
// MI455X (gfx1250) — matrix operand order repaired and checked by reading, not compiled, not run on hardware
//
#include <hip/hip_runtime.h>
#include <hip/hip_bf16.h>

typedef __attribute__((ext_vector_type(16))) _Float16 v16h;
typedef __attribute__((ext_vector_type(8)))  float    v8f;

union AFrag { v16h v; _Float16 h[16]; };
union CFrag { v8f  v; float    f[8];  };

// K index (within a 32-wide K step) held by A-fragment element e on `lane`
// per CDNA5 ISA 7.12.2 (16-bit A 16x32 layout).
__device__ __forceinline__ int a_kidx(int lane, int e) {
    return ((e >> 3) << 4) | ((lane >> 4) << 3) | (e & 7);
}

// ---------------------------------------------------------------------------
// Generic implicit-GEMM conv3d / convT3d (k=4, s=2, p=1, cubic power-of-2
// volumes; all extents passed as log2 so address math is shifts/masks only).
// mode 0: conv  (f16 weights [Cout][Cin][64]),       Din = 2*Dout
// mode 1: convT (f16 weights [Cin][CoutP][64]),      Din = Dout/2, sub-pixel
//         decomposition: M flattened (b, parity, od', oh', ow') so each
//         16-row tile is one parity class -> wave-uniform kernel taps.
// Activations: TIn/TOut are float or _Float16 (intermediates are f16).
// Weight Cout is padded to CoutP (multiple of 16) -> no masking in B loads.
// act: 0 none, 1 relu, 2 sigmoid
// ---------------------------------------------------------------------------
template <typename TIn, typename TOut>
__global__ __launch_bounds__(256)
void conv3d_wmma_kernel(const TIn* __restrict__ in,
                        const _Float16* __restrict__ wgt,
                        const float* __restrict__ bias,
                        TOut* __restrict__ out,
                        int Cin, int Cout, int CoutP, int lin, int ldo,
                        int mode, int act, int ntshift)
{
    const int lane  = threadIdx.x & 31;
    const int wave  = blockIdx.x * (blockDim.x >> 5) + (threadIdx.x >> 5);
    const int mtile = wave >> ntshift;
    const int ntile = wave & ((1 << ntshift) - 1);
    const int m0    = mtile << 4;
    const int mrow  = lane & 15;

    const int Din  = 1 << lin;
    const int lin2 = lin + lin;
    const int lin3 = lin2 + lin;

    // ---- tile coordinate decode (shifts/masks only) ----
    int b, base_d, base_h, base_w, spBase, spStep;
    int pd = 0, ph = 0, pw = 0;
    if (mode == 0) {
        const int mask = (1 << ldo) - 1;
        const int owb = m0 & mask;
        const int oh  = (m0 >> ldo) & mask;
        const int od  = (m0 >> (2 * ldo)) & mask;
        b = m0 >> (3 * ldo);
        base_d = od * 2 - 1;
        base_h = oh * 2 - 1;
        base_w = (owb + mrow) * 2 - 1;
        spBase = (((od << ldo) | oh) << ldo) | owb;
        spStep = 1;
    } else {
        const int lw2  = ldo - 1;
        const int mask = (1 << lw2) - 1;
        const int owpb = m0 & mask;
        const int ohp  = (m0 >> lw2) & mask;
        const int odp  = (m0 >> (2 * lw2)) & mask;
        const int par  = (m0 >> (3 * lw2)) & 7;
        b  = m0 >> (3 * lw2 + 3);
        pd = (par >> 2) & 1; ph = (par >> 1) & 1; pw = par & 1;
        base_d = odp + pd;
        base_h = ohp + ph;
        base_w = owpb + mrow + pw;
        const int od = odp * 2 + pd, oh = ohp * 2 + ph, ow = owpb * 2 + pw;
        spBase = (((od << ldo) | oh) << ldo) | ow;
        spStep = 2;
    }

    const int bBase = (b * Cin) << lin3;         // input batch base
    const int n0    = ntile << 4;                // first cout of this tile
    // convT precomputed kernel-tap base offset from parity:
    const int kBase = mode ? (((1 - pd) << 4) + ((1 - ph) << 2) + (1 - pw)) : 0;

    // ---- branchless A fragment (im2col rows) ----
    auto loadA = [&](int ks) {
        AFrag a;
        #pragma unroll
        for (int e = 0; e < 16; ++e) {
            const int k = (ks << 5) + a_kidx(lane, e);
            int id, ih, iw, cin;
            if (mode == 0) {
                cin = k >> 6;
                const int r = k & 63;
                id = base_d + (r >> 4);
                ih = base_h + ((r >> 2) & 3);
                iw = base_w + (r & 3);
            } else {
                cin = k >> 3;
                const int r = k & 7;
                id = base_d - ((r >> 2) & 1);
                ih = base_h - ((r >> 1) & 1);
                iw = base_w - (r & 1);
            }
            const bool valid = ((unsigned)id < (unsigned)Din) &
                               ((unsigned)ih < (unsigned)Din) &
                               ((unsigned)iw < (unsigned)Din);
            const int addr = bBase + (cin << lin3) + (id << lin2) + (ih << lin) + iw;
            const _Float16 v = (_Float16)in[valid ? addr : 0];
            a.h[e] = valid ? v : (_Float16)0.f;
        }
        return a;
    };

    // ---- B fragment (f16 weights, padded Cout -> no masking) ----
    auto loadB = [&](int ks) {
        AFrag wv;
        #pragma unroll
        for (int e = 0; e < 16; ++e) {
        const int k = (ks << 5) + a_kidx(lane, e);
        int wb, coStride;
        if (mode == 0) {
            const int cin = k >> 6, r = k & 63;      // r == kd*16+kh*4+kw
            wb = (cin << 6) + r;
            coStride = Cin << 6;
        } else {
            const int cin = k >> 3, r = k & 7;
            const int koff = kBase + (((r >> 2) & 1) << 5) +
                                     (((r >> 1) & 1) << 3) + ((r & 1) << 1);
            wb = cin * (CoutP << 6) + koff;
            coStride = 64;
        }
            wv.h[e] = wgt[wb + (n0 + mrow) * coStride];
        }
        return wv;
    };

    // ---- K loop: 2 WMMAs per iteration (nks even for all layers) ----
    const int nks = (mode ? (Cin >> 2) : (Cin << 1));
    CFrag acc = {};
    for (int ks = 0; ks < nks; ks += 2) {
        AFrag aa0 = loadA(ks);
        AFrag bb0 = loadB(ks);
        AFrag aa1 = loadA(ks + 1);
        AFrag bb1 = loadB(ks + 1);
        acc.v = __builtin_amdgcn_wmma_f32_16x16x32_f16(
            false, aa0.v, false, bb0.v, (short)0, acc.v, false, false);
        acc.v = __builtin_amdgcn_wmma_f32_16x16x32_f16(
            false, aa1.v, false, bb1.v, (short)0, acc.v, false, false);
    }

    // ---- epilogue: bias + activation + store (C layout: reg=row, lane=col) ----
    const int n    = lane & 15;
    const int cout = n0 + n;
    if (cout < Cout) {
        const int rowBase = (lane >> 4) << 3;
        const int outBase = (b * Cout + cout) << (3 * ldo);
        const float bv = bias[cout];
        #pragma unroll
        for (int r = 0; r < 8; ++r) {
            const int row = rowBase + r;
            float v = acc.f[r] + bv;
            if (act == 1)      v = v > 0.f ? v : 0.f;
            else if (act == 2) v = 1.f / (1.f + __expf(-v));
            out[outBase + spBase + row * spStep] = (TOut)v;
        }
    }
}

// ---------------------------------------------------------------------------
// f32 -> f16 conversion (layout preserved): weights + codebook
// ---------------------------------------------------------------------------
__global__ __launch_bounds__(256)
void cvt_f16_kernel(const float* __restrict__ src, _Float16* __restrict__ dst, int n)
{
    const int t = blockIdx.x * blockDim.x + threadIdx.x;
    if (t < n) dst[t] = (_Float16)src[t];
}

// dT3 weights [Cin=16][Cout=1][64] -> padded f16 [16][16][64], zeros for cout>0
__global__ __launch_bounds__(256)
void pad_convT_w_kernel(const float* __restrict__ src, _Float16* __restrict__ dst)
{
    const int t = blockIdx.x * blockDim.x + threadIdx.x;   // 16*16*64 = 16384
    if (t < 16 * 16 * 64) {
        const int cin  = t >> 10;
        const int cout = (t >> 6) & 15;
        const int j    = t & 63;
        dst[t] = cout == 0 ? (_Float16)src[(cin << 6) + j] : (_Float16)0.f;
    }
}

// ---------------------------------------------------------------------------
// VQ: codebook squared norms (f32 codebook for full-precision norms)
// ---------------------------------------------------------------------------
__global__ __launch_bounds__(256)
void vq_code_norms_kernel(const float* __restrict__ cb, float* __restrict__ cnorm)
{
    const int k = blockIdx.x * blockDim.x + threadIdx.x;
    if (k < 512) {
        float s = 0.f;
        #pragma unroll 8
        for (int c = 0; c < 64; ++c) { const float v = cb[k * 64 + c]; s += v * v; }
        cnorm[k] = s;
    }
}

// ---------------------------------------------------------------------------
// VQ argmin: per wave, 16 z-vectors vs all 512 codes via WMMA.
// dist' = ||c||^2 - 2<z,c> (||z||^2 constant per row, irrelevant for argmin)
// z_e layout: [B=4, C=64, sp=4096] -> zf rows are stride-4096 columns.
// Codebook consumed as preconverted f16 (pure d16 loads in the hot loop).
// ---------------------------------------------------------------------------
__global__ __launch_bounds__(256)
void vq_argmin_kernel(const float* __restrict__ z, const _Float16* __restrict__ cbh,
                      const float* __restrict__ cnorm, int* __restrict__ idxOut)
{
    const int lane = threadIdx.x & 31;
    const int wave = blockIdx.x * 8 + (threadIdx.x >> 5);
    const int m0   = wave << 4;
    const int mrow = m0 + (lane & 15);
    const int b    = mrow >> 12, sp = mrow & 4095;
    const float* zb = z + (b * 64) * 4096 + sp;

    AFrag a0, a1;                                   // K = 64 -> two K-steps
    #pragma unroll
    for (int e = 0; e < 16; ++e) {
        const int c = a_kidx(lane, e);
        a0.h[e] = (_Float16)zb[c * 4096];
        a1.h[e] = (_Float16)zb[(c + 32) * 4096];
    }

    float best[8]; int bidx[8];
    #pragma unroll
    for (int r = 0; r < 8; ++r) { best[r] = 3.4e38f; bidx[r] = 0; }

    const int n = lane & 15;
    for (int nt = 0; nt < 32; ++nt) {
        AFrag b0, b1;
        #pragma unroll
        for (int e = 0; e < 16; ++e) {
            const int code = (nt << 4) + n;
            const int c = a_kidx(lane, e);
            b0.h[e] = cbh[code * 64 + c];
            b1.h[e] = cbh[code * 64 + 32 + c];
        }
        CFrag acc = {};
        acc.v = __builtin_amdgcn_wmma_f32_16x16x32_f16(
            false, a0.v, false, b0.v, (short)0, acc.v, false, false);
        acc.v = __builtin_amdgcn_wmma_f32_16x16x32_f16(
            false, a1.v, false, b1.v, (short)0, acc.v, false, false);
        const int code = (nt << 4) + n;
        const float cn = cnorm[code];
        #pragma unroll
        for (int r = 0; r < 8; ++r) {
            const float d = cn - 2.f * acc.f[r];
            if (d < best[r]) { best[r] = d; bidx[r] = code; }
        }
    }
    // min-reduce across the 16 lanes that share the same rows (width-16 groups)
    #pragma unroll
    for (int mask = 1; mask < 16; mask <<= 1) {
        #pragma unroll
        for (int r = 0; r < 8; ++r) {
            const float ov = __shfl_xor(best[r], mask, 16);
            const int   oi = __shfl_xor(bidx[r], mask, 16);
            if (ov < best[r] || (ov == best[r] && oi < bidx[r])) {
                best[r] = ov; bidx[r] = oi;
            }
        }
    }
    if ((lane & 15) == 0) {
        const int rowBase = (lane >> 4) << 3;
        #pragma unroll
        for (int r = 0; r < 8; ++r)
            idxOut[m0 + rowBase + r] = bidx[r];
    }
}

// quantized[b,c,sp] = codebook[idx[b,sp], c]  (straight-through fwd value)
// Writes f32 tuple output AND a private f16 copy for the decoder.
__global__ __launch_bounds__(256)
void vq_gather_kernel(const int* __restrict__ idx, const float* __restrict__ cb,
                      float* __restrict__ q, _Float16* __restrict__ qh)
{
    const int t = blockIdx.x * blockDim.x + threadIdx.x;  // 4*64*4096 total
    if (t < 4 * 64 * 4096) {
        const int sp = t & 4095;
        const int c  = (t >> 12) & 63;
        const int b  = t >> 18;
        const float v = cb[idx[(b << 12) + sp] * 64 + c];
        q[t]  = v;
        qh[t] = (_Float16)v;
    }
}

// ---------------------------------------------------------------------------
extern "C" void kernel_launch(void* const* d_in, const int* in_sizes, int n_in,
                              void* d_out, int out_size, void* d_ws, size_t ws_size,
                              hipStream_t stream)
{
    const float* x   = (const float*)d_in[0];
    const float* w1  = (const float*)d_in[1];  const float* b1  = (const float*)d_in[2];
    const float* w2  = (const float*)d_in[3];  const float* b2  = (const float*)d_in[4];
    const float* w3  = (const float*)d_in[5];  const float* b3  = (const float*)d_in[6];
    const float* cb  = (const float*)d_in[7];
    const float* dw1 = (const float*)d_in[8];  const float* db1 = (const float*)d_in[9];
    const float* dw2 = (const float*)d_in[10]; const float* db2 = (const float*)d_in[11];
    const float* dw3 = (const float*)d_in[12]; const float* db3 = (const float*)d_in[13];

    float* out  = (float*)d_out;
    float* xhat = out;                 // 4*1*128^3   = 8388608  (f32)
    float* q    = out + 8388608;       // 4*64*16^3   = 1048576  (f32)
    float* ze   = out + 9437184;       // 4*64*16^3   = 1048576  (f32)

    // ---- workspace layout (f16 intermediates halve HBM traffic) ----
    char* ws = (char*)d_ws;
    _Float16* bufA  = (_Float16*)(ws);              // 16,777,216 h: h1 / g2 (32 MB)
    _Float16* bufB  = (_Float16*)(ws + 33554432);   //  4,194,304 h: h2 / g1 (8 MB)
    _Float16* qh    = (_Float16*)(ws + 41943040);   //  1,048,576 h: quantized f16
    _Float16* w1h   = (_Float16*)(ws + 44040192);   //   1,024 h
    _Float16* w2h   = (_Float16*)(ws + 44042240);   //  32,768 h
    _Float16* w3h   = (_Float16*)(ws + 44107776);   // 131,072 h
    _Float16* dw1h  = (_Float16*)(ws + 44369920);   // 131,072 h
    _Float16* dw2h  = (_Float16*)(ws + 44632064);   //  32,768 h
    _Float16* dw3h  = (_Float16*)(ws + 44697600);   //  16,384 h (Cout padded 1->16)
    float*    cnorm = (float*)   (ws + 44730368);   //     512 f
    int*      vidx  = (int*)     (ws + 44732416);   //  16,384 i
    _Float16* cbh   = (_Float16*)(ws + 44797952);   //  32,768 h: codebook f16

    // ---- weight + codebook conversion to f16 (tiny, once per call) ----
    cvt_f16_kernel<<<4,   256, 0, stream>>>(w1,  w1h,  1024);
    cvt_f16_kernel<<<128, 256, 0, stream>>>(w2,  w2h,  32768);
    cvt_f16_kernel<<<512, 256, 0, stream>>>(w3,  w3h,  131072);
    cvt_f16_kernel<<<512, 256, 0, stream>>>(dw1, dw1h, 131072);
    cvt_f16_kernel<<<128, 256, 0, stream>>>(dw2, dw2h, 32768);
    pad_convT_w_kernel<<<64, 256, 0, stream>>>(dw3, dw3h);
    cvt_f16_kernel<<<128, 256, 0, stream>>>(cb, cbh, 32768);

    // ---- encoder ----  (Cin, Cout, CoutP, log2 Din, log2 Dout, mode, act, nt)
    // conv1: 1->16, 128->64, relu, f32 in / f16 out.  65536 waves
    conv3d_wmma_kernel<float, _Float16><<<8192, 256, 0, stream>>>(
        x, w1h, b1, bufA, 1, 16, 16, 7, 6, 0, 1, 0);
    // conv2: 16->32, 64->32, relu, f16/f16.  16384 waves
    conv3d_wmma_kernel<_Float16, _Float16><<<2048, 256, 0, stream>>>(
        bufA, w2h, b2, bufB, 16, 32, 32, 6, 5, 0, 1, 1);
    // conv3: 32->64, 32->16, no act -> z_e f32.  4096 waves
    conv3d_wmma_kernel<_Float16, float><<<512, 256, 0, stream>>>(
        bufB, w3h, b3, ze, 32, 64, 64, 5, 4, 0, 0, 2);

    // ---- vector quantization ----
    vq_code_norms_kernel<<<2, 256, 0, stream>>>(cb, cnorm);
    vq_argmin_kernel<<<128, 256, 0, stream>>>(ze, cbh, cnorm, vidx);
    vq_gather_kernel<<<4096, 256, 0, stream>>>(vidx, cb, q, qh);

    // ---- decoder (convT as 8 parity-class k=2 convs) ----
    // dT1: 64->32, 16->32, relu, f16/f16.  16384 waves
    conv3d_wmma_kernel<_Float16, _Float16><<<2048, 256, 0, stream>>>(
        qh, dw1h, db1, bufB, 64, 32, 32, 4, 5, 1, 1, 1);
    // dT2: 32->16, 32->64, relu, f16/f16.  65536 waves
    conv3d_wmma_kernel<_Float16, _Float16><<<8192, 256, 0, stream>>>(
        bufB, dw2h, db2, bufA, 32, 16, 16, 5, 6, 1, 1, 0);
    // dT3: 16->1 (padded 16), 64->128, sigmoid, f16 in / f32 out. 524288 waves
    conv3d_wmma_kernel<_Float16, float><<<65536, 256, 0, stream>>>(
        bufA, dw3h, db3, xhat, 16, 1, 16, 6, 7, 1, 2, 0);
}
